// MessagePassingLayer_51101520887961
// MI455X (gfx1250) — matrix operand order repaired and checked by reading, not compiled, not run on hardware
//
#include <hip/hip_runtime.h>
#include <hip/hip_bf16.h>

// ---------------------------------------------------------------------------
// MessagePassing GNN layer on gfx1250 (MI455X): bf16 WMMA GEMMs, fp32 accum.
// ---------------------------------------------------------------------------

typedef __attribute__((ext_vector_type(16))) __bf16 v16bf;
typedef __attribute__((ext_vector_type(8)))  __bf16 v8bf;
typedef __attribute__((ext_vector_type(4)))  __bf16 v4bf;
typedef __attribute__((ext_vector_type(8)))  float  v8f;

#define HDIM   128
#define NT     8          // number of 16-wide N tiles in 128 output cols
#define KPAD1  288        // 260 padded to multiple of 32
#define KF1    9          // 288/32 K-fragments, edge GEMM1
#define KF2    4          // 128/32, GEMM2
#define KFU1   8          // 256/32, node GEMM1

// ---------------------------------------------------------------------------
// Pack fp32 row-major weight [K x 128] into bf16 WMMA B-fragment order.
// B-matrix layout (ISA 7.12.2, 16-bit, 32x16 tile): lane n (0..15) holds
// K = kb..kb+7, kb+16..kb+23; lane n+16 holds K = kb+8..kb+15, kb+24..kb+31.
// Fragment storage: frag (kf, nf) -> 32 lanes x 16 bf16 contiguous.
// ---------------------------------------------------------------------------
__global__ void pack_weights_kernel(const float* __restrict__ W,
                                    __bf16* __restrict__ dst,
                                    int Kvalid, int KF)
{
    int idx   = blockIdx.x * blockDim.x + threadIdx.x;
    int total = KF * NT * 32;
    if (idx >= total) return;
    int lane = idx & 31;
    int nf   = (idx >> 5) & (NT - 1);
    int kf   = idx / (32 * NT);
    int n    = nf * 16 + (lane & 15);
    int kb   = kf * 32 + ((lane >> 4) ? 8 : 0);
    __bf16* o = dst + (size_t)idx * 16;
#pragma unroll
    for (int i = 0; i < 16; ++i) {
        int k   = kb + (i < 8 ? i : i + 8);
        float v = (k < Kvalid) ? W[(size_t)k * HDIM + n] : 0.0f;
        o[i] = (__bf16)v;
    }
}

__global__ void zero_f32_kernel(float* __restrict__ p, size_t n)
{
    size_t i = (size_t)blockIdx.x * blockDim.x + threadIdx.x;
    size_t stride = (size_t)gridDim.x * blockDim.x;
    for (; i < n; i += stride) p[i] = 0.0f;
}

// ---------------------------------------------------------------------------
// Load A fragment (16x32 bf16, ISA 7.12.2) from a row-major bf16 LDS tile.
// Lane (m = lane%16) reads two contiguous 16B chunks:
//   lanes 0..15 : K = kb..kb+7  and kb+16..kb+23
//   lanes 16..31: K = kb+8..kb+15 and kb+24..kb+31
// ---------------------------------------------------------------------------
__device__ __forceinline__ v16bf load_a_frag(const __bf16* tile, int stride,
                                             int lane, int kbase)
{
    int m = lane & 15;
    int h = (lane >> 4) * 8;
    const __bf16* row = tile + m * stride + kbase + h;
    v8bf lo = *reinterpret_cast<const v8bf*>(row);
    v8bf hi = *reinterpret_cast<const v8bf*>(row + 16);
    return __builtin_shufflevector(lo, hi, 0, 1, 2, 3, 4, 5, 6, 7,
                                           8, 9, 10, 11, 12, 13, 14, 15);
}

// SiLU via hardware reciprocal (v_rcp_f32) instead of IEEE divide.
__device__ __forceinline__ float silu(float x)
{
    float e = __expf(-x);
    return x * __builtin_amdgcn_rcpf(1.0f + e);
}

// Convert float4 -> 4 packed bf16.
__device__ __forceinline__ v4bf cvt4(float4 v)
{
    v4bf b;
    b[0] = (__bf16)v.x; b[1] = (__bf16)v.y;
    b[2] = (__bf16)v.z; b[3] = (__bf16)v.w;
    return b;
}

// ---------------------------------------------------------------------------
// Edge kernel: per 16-edge tile, msg = (silu([hs|hd|e] @ W1 + b1)) @ W2 + b2,
// then fp32 atomic scatter-add into agg[dst]. 8 waves; wave w owns N-tile w.
// ---------------------------------------------------------------------------
__global__ void __launch_bounds__(256)
edge_kernel(const float* __restrict__ nodef,
            const float* __restrict__ edgef,
            const int*   __restrict__ eidx,       // [2, E]
            const __bf16* __restrict__ W1p, const float* __restrict__ b1,
            const __bf16* __restrict__ W2p, const float* __restrict__ b2,
            float* __restrict__ agg, int ntiles, int Etot)
{
    __shared__ __bf16 a_lds[16 * KPAD1];          // 16 x 288 message inputs
    __shared__ __bf16 h_lds[16 * HDIM];           // 16 x 128 hidden
    __shared__ int    dst_lds[16];

    const int tid   = threadIdx.x;
    const int lane  = tid & 31;
    const int w     = tid >> 5;        // wave id == N tile id
    const int mcol  = lane & 15;       // C-frag column within tile
    const int rbase = (lane >> 4) * 8; // C-frag row base

    // Loop-invariant weight fragments live in VGPRs for the whole kernel.
    v16bf w1f[KF1];
#pragma unroll
    for (int kf = 0; kf < KF1; ++kf)
        w1f[kf] = *reinterpret_cast<const v16bf*>(
            W1p + (((size_t)kf * NT + w) * 32 + lane) * 16);
    v16bf w2f[KF2];
#pragma unroll
    for (int kf = 0; kf < KF2; ++kf)
        w2f[kf] = *reinterpret_cast<const v16bf*>(
            W2p + (((size_t)kf * NT + w) * 32 + lane) * 16);
    const float bias1 = b1[w * 16 + mcol];
    const float bias2 = b2[w * 16 + mcol];

    for (int tile = blockIdx.x; tile < ntiles; tile += gridDim.x) {
        const int e0 = tile * 16;

        // ---- stage A tile: [src_h(128) | dst_h(128) | edge(4) | pad(28)] ----
        // Vectorized: float4 gathers -> packed bf16 -> 8B LDS stores.
        {
            int row = tid >> 4;                 // 0..15
            int c16 = tid & 15;
            int e   = e0 + row;
            int ec  = (e < Etot) ? e : (Etot - 1);
            int s   = eidx[ec];
            int d   = eidx[Etot + ec];
            if (c16 == 0) dst_lds[row] = d;
            const float4* sp4 = reinterpret_cast<const float4*>(nodef + (size_t)s * HDIM);
            const float4* dp4 = reinterpret_cast<const float4*>(nodef + (size_t)d * HDIM);
            __bf16* arow = a_lds + row * KPAD1;
#pragma unroll
            for (int cc = 0; cc < 2; ++cc) {
                int c = c16 + cc * 16;          // float4 chunk 0..31
                float4 sv = sp4[c];
                float4 dv = dp4[c];
                *reinterpret_cast<v4bf*>(arow + c * 4)        = cvt4(sv);
                *reinterpret_cast<v4bf*>(arow + HDIM + c * 4) = cvt4(dv);
            }
            // tail: 32 elements = 8 chunks of 4; chunk 0 = edge feats, rest 0
            if (c16 < 8) {
                v4bf t;
                if (c16 == 0) {
                    const float4 ev = *reinterpret_cast<const float4*>(edgef + (size_t)ec * 4);
                    t = cvt4(ev);
                } else {
                    t[0] = (__bf16)0.0f; t[1] = (__bf16)0.0f;
                    t[2] = (__bf16)0.0f; t[3] = (__bf16)0.0f;
                }
                *reinterpret_cast<v4bf*>(arow + 2 * HDIM + c16 * 4) = t;
            }
        }
        __syncthreads();

        // ---- GEMM1: 16x16 tile over K=288, one-fragment A prefetch ----
        v8f acc;
#pragma unroll
        for (int i = 0; i < 8; ++i) acc[i] = bias1;
        v16bf a_cur = load_a_frag(a_lds, KPAD1, lane, 0);
#pragma unroll
        for (int kf = 0; kf < KF1; ++kf) {
            v16bf a_nxt = (kf + 1 < KF1)
                        ? load_a_frag(a_lds, KPAD1, lane, (kf + 1) * 32)
                        : a_cur;
            acc = __builtin_amdgcn_wmma_f32_16x16x32_bf16(
                false, a_cur, false, w1f[kf], (short)0, acc, false, false);
            a_cur = a_nxt;
        }

        // ---- SiLU, stash hidden tile for GEMM2 ----
#pragma unroll
        for (int r = 0; r < 8; ++r) {
            float v = silu(acc[r]);
            h_lds[(rbase + r) * HDIM + w * 16 + mcol] = (__bf16)v;
        }
        __syncthreads();

        // ---- GEMM2: K=128 ----
        v8f acc2;
#pragma unroll
        for (int i = 0; i < 8; ++i) acc2[i] = bias2;
        v16bf h_cur = load_a_frag(h_lds, HDIM, lane, 0);
#pragma unroll
        for (int kf = 0; kf < KF2; ++kf) {
            v16bf h_nxt = (kf + 1 < KF2)
                        ? load_a_frag(h_lds, HDIM, lane, (kf + 1) * 32)
                        : h_cur;
            acc2 = __builtin_amdgcn_wmma_f32_16x16x32_bf16(
                false, h_cur, false, w2f[kf], (short)0, acc2, false, false);
            h_cur = h_nxt;
        }

        // ---- scatter-add messages into aggregate ----
#pragma unroll
        for (int r = 0; r < 8; ++r) {
            int row = rbase + r;
            if (e0 + row < Etot) {
                int d = dst_lds[row];
                unsafeAtomicAdd(&agg[(size_t)d * HDIM + w * 16 + mcol], acc2[r]);
            }
        }
        __syncthreads();   // protect LDS reuse next iteration
    }
}

// ---------------------------------------------------------------------------
// Node kernel: upd = silu([x|agg] @ U1 + b1) @ U2 + b2; LayerNorm(x + upd).
// ---------------------------------------------------------------------------
__global__ void __launch_bounds__(256)
node_kernel(const float* __restrict__ nodef,
            const float* __restrict__ agg,
            const __bf16* __restrict__ U1p, const float* __restrict__ b1,
            const __bf16* __restrict__ U2p, const float* __restrict__ b2,
            const float* __restrict__ ln_g, const float* __restrict__ ln_b,
            float* __restrict__ out, int ntiles, int Ntot)
{
    __shared__ __bf16 a_lds[16 * 256];
    __shared__ __bf16 h_lds[16 * HDIM];
    __shared__ float  o_lds[16 * 132];   // padded stride vs bank conflicts

    const int tid   = threadIdx.x;
    const int lane  = tid & 31;
    const int w     = tid >> 5;
    const int mcol  = lane & 15;
    const int rbase = (lane >> 4) * 8;

    v16bf u1f[KFU1];
#pragma unroll
    for (int kf = 0; kf < KFU1; ++kf)
        u1f[kf] = *reinterpret_cast<const v16bf*>(
            U1p + (((size_t)kf * NT + w) * 32 + lane) * 16);
    v16bf u2f[KF2];
#pragma unroll
    for (int kf = 0; kf < KF2; ++kf)
        u2f[kf] = *reinterpret_cast<const v16bf*>(
            U2p + (((size_t)kf * NT + w) * 32 + lane) * 16);
    const float bias1 = b1[w * 16 + mcol];
    const float bias2 = b2[w * 16 + mcol];

    for (int tile = blockIdx.x; tile < ntiles; tile += gridDim.x) {
        const int n0 = tile * 16;

        // ---- stage [x | agg] tile, float4 loads -> packed bf16 ----
        {
            int row = tid >> 4;
            int c16 = tid & 15;
            int n   = n0 + row;
            int nc  = (n < Ntot) ? n : (Ntot - 1);
            const float4* xp4 = reinterpret_cast<const float4*>(nodef + (size_t)nc * HDIM);
            const float4* ap4 = reinterpret_cast<const float4*>(agg   + (size_t)nc * HDIM);
            __bf16* arow = a_lds + row * 256;
#pragma unroll
            for (int cc = 0; cc < 2; ++cc) {
                int c = c16 + cc * 16;
                float4 xv = xp4[c];
                float4 av = ap4[c];
                *reinterpret_cast<v4bf*>(arow + c * 4)        = cvt4(xv);
                *reinterpret_cast<v4bf*>(arow + HDIM + c * 4) = cvt4(av);
            }
        }
        __syncthreads();

        // ---- GEMM1: K=256 ----
        v8f acc;
#pragma unroll
        for (int i = 0; i < 8; ++i) acc[i] = bias1;
        v16bf a_cur = load_a_frag(a_lds, 256, lane, 0);
#pragma unroll
        for (int kf = 0; kf < KFU1; ++kf) {
            v16bf a_nxt = (kf + 1 < KFU1)
                        ? load_a_frag(a_lds, 256, lane, (kf + 1) * 32)
                        : a_cur;
            acc = __builtin_amdgcn_wmma_f32_16x16x32_bf16(
                false, a_cur, false, u1f[kf], (short)0, acc, false, false);
            a_cur = a_nxt;
        }
#pragma unroll
        for (int r = 0; r < 8; ++r) {
            float v = silu(acc[r]);
            h_lds[(rbase + r) * HDIM + w * 16 + mcol] = (__bf16)v;
        }
        __syncthreads();

        // ---- GEMM2: K=128, + bias + residual into fp32 LDS tile ----
        v8f acc2;
#pragma unroll
        for (int i = 0; i < 8; ++i) acc2[i] = bias2;
        v16bf h_cur = load_a_frag(h_lds, HDIM, lane, 0);
#pragma unroll
        for (int kf = 0; kf < KF2; ++kf) {
            v16bf h_nxt = (kf + 1 < KF2)
                        ? load_a_frag(h_lds, HDIM, lane, (kf + 1) * 32)
                        : h_cur;
            acc2 = __builtin_amdgcn_wmma_f32_16x16x32_bf16(
                false, h_cur, false, u2f[kf], (short)0, acc2, false, false);
            h_cur = h_nxt;
        }
#pragma unroll
        for (int r = 0; r < 8; ++r) {
            int row = rbase + r;
            int n   = n0 + row;
            int nc  = (n < Ntot) ? n : (Ntot - 1);
            float res = nodef[(size_t)nc * HDIM + w * 16 + mcol];
            o_lds[row * 132 + w * 16 + mcol] = acc2[r] + res;
        }
        __syncthreads();

        // ---- LayerNorm: wave w handles rows 2w, 2w+1 (4 elems per lane) ----
#pragma unroll
        for (int rr = 0; rr < 2; ++rr) {
            int row = w * 2 + rr;
            float vals[4];
            float s = 0.0f, ss = 0.0f;
#pragma unroll
            for (int j = 0; j < 4; ++j) {
                float v = o_lds[row * 132 + lane * 4 + j];
                vals[j] = v;
                s  += v;
                ss += v * v;
            }
#pragma unroll
            for (int m = 16; m >= 1; m >>= 1) {
                s  += __shfl_xor(s,  m, 32);
                ss += __shfl_xor(ss, m, 32);
            }
            float mu  = s * (1.0f / 128.0f);
            float var = ss * (1.0f / 128.0f) - mu * mu;
            float inv = rsqrtf(var + 1e-5f);
            int n = n0 + row;
            if (n < Ntot) {
                const float4 g4 = *reinterpret_cast<const float4*>(ln_g + lane * 4);
                const float4 b4 = *reinterpret_cast<const float4*>(ln_b + lane * 4);
                float4 o;
                o.x = (vals[0] - mu) * inv * g4.x + b4.x;
                o.y = (vals[1] - mu) * inv * g4.y + b4.y;
                o.z = (vals[2] - mu) * inv * g4.z + b4.z;
                o.w = (vals[3] - mu) * inv * g4.w + b4.w;
                *reinterpret_cast<float4*>(out + (size_t)n * HDIM + lane * 4) = o;
            }
        }
        __syncthreads();
    }
}

// ---------------------------------------------------------------------------
// Host launcher
// ---------------------------------------------------------------------------
extern "C" void kernel_launch(void* const* d_in, const int* in_sizes, int n_in,
                              void* d_out, int out_size, void* d_ws, size_t ws_size,
                              hipStream_t stream)
{
    const float* nodef = (const float*)d_in[0];   // [N,128]
    const float* edgef = (const float*)d_in[1];   // [E,4]
    const float* mW1   = (const float*)d_in[2];   // [260,128]
    const float* mb1   = (const float*)d_in[3];
    const float* mW2   = (const float*)d_in[4];   // [128,128]
    const float* mb2   = (const float*)d_in[5];
    const float* uW1   = (const float*)d_in[6];   // [256,128]
    const float* ub1   = (const float*)d_in[7];
    const float* uW2   = (const float*)d_in[8];   // [128,128]
    const float* ub2   = (const float*)d_in[9];
    const float* ln_g  = (const float*)d_in[10];
    const float* ln_b  = (const float*)d_in[11];
    const int*   eidx  = (const int*)d_in[12];    // [2,E]

    const int N = in_sizes[0] / HDIM;
    const int E = in_sizes[12] / 2;

    // Workspace layout
    char* ws = (char*)d_ws;
    size_t off = 0;
    float* agg = (float*)(ws + off);
    off += (size_t)N * HDIM * sizeof(float);
    off = (off + 255) & ~(size_t)255;
    __bf16* W1p = (__bf16*)(ws + off); off += (size_t)KF1  * NT * 32 * 16 * 2;
    off = (off + 255) & ~(size_t)255;
    __bf16* W2p = (__bf16*)(ws + off); off += (size_t)KF2  * NT * 32 * 16 * 2;
    off = (off + 255) & ~(size_t)255;
    __bf16* U1p = (__bf16*)(ws + off); off += (size_t)KFU1 * NT * 32 * 16 * 2;
    off = (off + 255) & ~(size_t)255;
    __bf16* U2p = (__bf16*)(ws + off);

    // 1) zero the aggregation buffer
    {
        size_t n = (size_t)N * HDIM;
        zero_f32_kernel<<<512, 256, 0, stream>>>(agg, n);
    }
    // 2) pack weights to bf16 fragment order
    {
        int t1 = KF1 * NT * 32, t2 = KF2 * NT * 32, tu1 = KFU1 * NT * 32;
        pack_weights_kernel<<<(t1 + 255) / 256, 256, 0, stream>>>(mW1, W1p, 260, KF1);
        pack_weights_kernel<<<(t2 + 255) / 256, 256, 0, stream>>>(mW2, W2p, 128, KF2);
        pack_weights_kernel<<<(tu1 + 255) / 256, 256, 0, stream>>>(uW1, U1p, 256, KFU1);
        pack_weights_kernel<<<(t2 + 255) / 256, 256, 0, stream>>>(uW2, U2p, 128, KF2);
    }
    // 3) edge message + scatter
    {
        int ntiles = (E + 15) / 16;
        int blocks = ntiles < 2048 ? ntiles : 2048;
        edge_kernel<<<blocks, 256, 0, stream>>>(nodef, edgef, eidx,
                                                W1p, mb1, W2p, mb2,
                                                agg, ntiles, E);
    }
    // 4) node update + LayerNorm
    {
        int ntiles = (N + 15) / 16;
        int blocks = ntiles < 1024 ? ntiles : 1024;
        node_kernel<<<blocks, 256, 0, stream>>>(nodef, agg,
                                                U1p, ub1, U2p, ub2,
                                                ln_g, ln_b,
                                                (float*)d_out, ntiles, N);
    }
}
